// MNISTRNN_63866163691710
// MI455X (gfx1250) — matrix operand order repaired and checked by reading, not compiled, not run on hardware
//
#include <hip/hip_runtime.h>
#include <hip/hip_bf16.h>

// ---------------------------------------------------------------------------
// MNIST-RNN fused step for MI455X (gfx1250, wave32, WMMA).
// Precision: f16 A/B operands, f32 WMMA accumulation (inputs are bounded:
// image in [0,1], h0 ~ N(0,1), weights ~ 0.05*N(0,1) -> f16 mantissa is ample).
// Data path: crops gathered directly into LDS (no HBM round trip for x),
// f16 weights converted once and served from L2/WGP$ to every block.
// ---------------------------------------------------------------------------

typedef __attribute__((ext_vector_type(16))) _Float16 v16h;
typedef __attribute__((ext_vector_type(8)))  _Float16 v8h;
typedef __attribute__((ext_vector_type(8)))  float    v8f;

#define CROPSZ 20
#define HALFW  10
#define HID    128
#define KPAD   416   // 400 padded to a multiple of 32 (WMMA K)
#define BM     64    // batch rows per block (4 waves x 16-row M slabs)
#define XSTR   432   // LDS row stride in halfs (216 dwords -> spreads banks)
#define NTILES 8     // 128 / 16 output-feature tiles

// Build the f16 A-fragment for v_wmma_f32_16x16x32_f16.
// A 16x32 layout (matching B): lanes 0-15 hold M=lane, K 0..15;
// lanes 16-31 hold M=lane-16, K 16..31. Per lane that is 16 contiguous
// halfs -> two adjacent b128 loads.
__device__ __forceinline__ v16h make_afrag(const _Float16* p) {
  v8h lo = *(const v8h*)(p);
  v8h hi = *(const v8h*)(p + 8);
  return __builtin_shufflevector(lo, hi,
      0, 1, 2, 3, 4, 5, 6, 7, 8, 9, 10, 11, 12, 13, 14, 15);
}

// B 32x16 layout: lane n holds 16 contiguous K values of column n
// (lanes 16-31: K+16). Column n of B == row n of W (row-major), so this is
// a single 32B contiguous load per lane.
__device__ __forceinline__ v8f wmma_acc(v16h a, const _Float16* bp, v8f c) {
  v16h b = *(const v16h*)bp;
  return __builtin_amdgcn_wmma_f32_16x16x32_f16(
      /*neg_a=*/false, a, /*neg_b=*/false, b,
      /*c_mod=*/(short)0, c, /*reuse_a=*/false, /*reuse_b=*/false);
}

// ---------------------------------------------------------------------------
// Prep: convert weights to f16 (K-padded), fuse heads, pre-sum biases.
// ---------------------------------------------------------------------------
__global__ void prep_params(const float* Wih0, const float* Whh0,
                            const float* Wih1, const float* Whh1,
                            const float* Wcls, const float* Wact,
                            const float* bih0, const float* bhh0,
                            const float* bih1, const float* bhh1,
                            const float* bcls, const float* bact,
                            _Float16* Wih0h, _Float16* Whh0h,
                            _Float16* Wih1h, _Float16* Whh1h,
                            _Float16* Wheadh, float* bias0, float* bias1,
                            float* bhead) {
  int gid = blockIdx.x * blockDim.x + threadIdx.x;
  int stride = gridDim.x * blockDim.x;
  for (int i = gid; i < HID * KPAD; i += stride) {
    int r = i / KPAD, k = i % KPAD;
    Wih0h[i] = (_Float16)(k < 400 ? Wih0[r * 400 + k] : 0.f);
  }
  for (int i = gid; i < HID * HID; i += stride) {
    Whh0h[i] = (_Float16)Whh0[i];
    Wih1h[i] = (_Float16)Wih1[i];
    Whh1h[i] = (_Float16)Whh1[i];
  }
  for (int i = gid; i < 16 * HID; i += stride) {
    int r = i / HID, k = i % HID;
    float v = (r < 10) ? Wcls[r * HID + k]
                       : ((r < 12) ? Wact[(r - 10) * HID + k] : 0.f);
    Wheadh[i] = (_Float16)v;
  }
  for (int i = gid; i < HID; i += stride) {
    bias0[i] = bih0[i] + bhh0[i];
    bias1[i] = bih1[i] + bhh1[i];
  }
  for (int i = gid; i < 16; i += stride)
    bhead[i] = (i < 10) ? bcls[i] : ((i < 12) ? bact[i - 10] : 0.f);
}

__global__ void prep_h0(const float* h0, _Float16* h0h, int n) {
  int i = blockIdx.x * blockDim.x + threadIdx.x;
  int stride = gridDim.x * blockDim.x;
  for (; i < n; i += stride) h0h[i] = (_Float16)h0[i];
}

// ---------------------------------------------------------------------------
// Fused main kernel: crop->LDS, layer0 GEMM+tanh, layer1 GEMM+tanh, heads.
// Block = 128 threads (4 wave32), BM=64 batch rows, each wave owns a 16-row
// M slab end-to-end (so LDS h-state rows are wave-private: one barrier only).
// ---------------------------------------------------------------------------
__global__ __launch_bounds__(128) void rnn_fused(
    const float* __restrict__ image, const int* __restrict__ center,
    const _Float16* __restrict__ h0h,       // (2,B,128) f16
    const _Float16* __restrict__ Wih0h,     // (128,KPAD)
    const _Float16* __restrict__ Whh0h,     // (128,128)
    const _Float16* __restrict__ Wih1h,     // (128,128)
    const _Float16* __restrict__ Whh1h,     // (128,128)
    const _Float16* __restrict__ Wheadh,    // (16,128)
    const float* __restrict__ bias0, const float* __restrict__ bias1,
    const float* __restrict__ bhead, float* __restrict__ out, int Btot) {
  __shared__ __align__(32) _Float16 xs[BM * XSTR];

  const int tid = threadIdx.x;
  const int lane = tid & 31;
  const int wave = tid >> 5;
  const int blockM = blockIdx.x * BM;

  // ---- Phase 0: gather 20x20 zero-padded crops into LDS as f16 ----------
  for (int j = tid; j < BM * CROPSZ; j += blockDim.x) {
    int s = j / CROPSZ, r = j % CROPSZ;
    int b = blockM + s;
    int cx = center[2 * b + 0];
    int cy = center[2 * b + 1];
    int orow = cy - 2 * HALFW + r;       // row in original 28x28 image
    _Float16* dst = &xs[s * XSTR + r * CROPSZ];
    if (orow >= 0 && orow < 28) {
      const float* src = image + (size_t)b * 784 + (size_t)orow * 28;
      int oc0 = cx - 2 * HALFW;
#pragma unroll
      for (int c = 0; c < CROPSZ; ++c) {
        int oc = oc0 + c;
        float v = (oc >= 0 && oc < 28) ? src[oc] : 0.f;
        dst[c] = (_Float16)v;
      }
    } else {
#pragma unroll
      for (int c = 0; c < CROPSZ; ++c) dst[c] = (_Float16)0.f;
    }
  }
  // zero the K padding tail (cols 400..431)
  for (int j = tid; j < BM * (XSTR - 400); j += blockDim.x) {
    int s = j / (XSTR - 400), c = j % (XSTR - 400);
    xs[s * XSTR + 400 + c] = (_Float16)0.f;
  }
  __syncthreads();

  const int mloc16 = lane & 15;                 // M within slab for A rows
  const int mrow = (wave << 4) + mloc16;        // local A row for this lane
  const int aKoff = (lane >> 4) << 4;           // +16 halfs for upper half-wave
  const int bKoff = (lane >> 4) << 4;           // +16 halfs for B upper half
  const int nloc = lane & 15;                   // N within tile (C/D layout)
  const int mAdd = (lane >> 4) << 3;            // +8 M rows for upper lanes
  const size_t hnBase = (size_t)Btot * 12;      // after class(10)+action(2)

  v8f acc[NTILES];
#pragma unroll
  for (int nt = 0; nt < NTILES; ++nt)
#pragma unroll
    for (int i = 0; i < 8; ++i) acc[nt][i] = 0.f;

  // ---- Layer 0: x(Kpad=416) @ Wih0^T  +  h0[0](K=128) @ Whh0^T ----------
  for (int k0 = 0; k0 < KPAD; k0 += 32) {
    v16h a = make_afrag(&xs[mrow * XSTR + k0 + aKoff]);
#pragma unroll
    for (int nt = 0; nt < NTILES; ++nt)
      acc[nt] = wmma_acc(a, &Wih0h[(nt * 16 + nloc) * KPAD + k0 + bKoff],
                         acc[nt]);
  }
  const _Float16* h0a = h0h + (size_t)(blockM + mrow) * HID;
  for (int k0 = 0; k0 < HID; k0 += 32) {
    v16h a = make_afrag(h0a + k0 + aKoff);
#pragma unroll
    for (int nt = 0; nt < NTILES; ++nt)
      acc[nt] = wmma_acc(a, &Whh0h[(nt * 16 + nloc) * HID + k0 + bKoff],
                         acc[nt]);
  }
  // epilogue: tanh, write h1 to hn[0], stash f16 h1 in this wave's LDS rows
#pragma unroll
  for (int nt = 0; nt < NTILES; ++nt) {
    int n = nt * 16 + nloc;
    float bv = bias0[n];
#pragma unroll
    for (int r = 0; r < 8; ++r) {
      int m = (wave << 4) + r + mAdd;
      float v = tanhf(acc[nt][r] + bv);
      out[hnBase + (size_t)(blockM + m) * HID + n] = v;
      xs[m * XSTR + n] = (_Float16)v;   // wave-private rows: no barrier
      acc[nt][r] = 0.f;                 // reset for layer 1
    }
  }

  // ---- Layer 1: h1(K=128) @ Wih1^T  +  h0[1](K=128) @ Whh1^T ------------
  for (int k0 = 0; k0 < HID; k0 += 32) {
    v16h a = make_afrag(&xs[mrow * XSTR + k0 + aKoff]);
#pragma unroll
    for (int nt = 0; nt < NTILES; ++nt)
      acc[nt] = wmma_acc(a, &Wih1h[(nt * 16 + nloc) * HID + k0 + bKoff],
                         acc[nt]);
  }
  const _Float16* h0b = h0h + (size_t)Btot * HID + (size_t)(blockM + mrow) * HID;
  for (int k0 = 0; k0 < HID; k0 += 32) {
    v16h a = make_afrag(h0b + k0 + aKoff);
#pragma unroll
    for (int nt = 0; nt < NTILES; ++nt)
      acc[nt] = wmma_acc(a, &Whh1h[(nt * 16 + nloc) * HID + k0 + bKoff],
                         acc[nt]);
  }
  // epilogue: tanh, write h2 to hn[1], overwrite LDS rows with f16 h2
#pragma unroll
  for (int nt = 0; nt < NTILES; ++nt) {
    int n = nt * 16 + nloc;
    float bv = bias1[n];
#pragma unroll
    for (int r = 0; r < 8; ++r) {
      int m = (wave << 4) + r + mAdd;
      float v = tanhf(acc[nt][r] + bv);
      out[hnBase + (size_t)Btot * HID + (size_t)(blockM + m) * HID + n] = v;
      xs[m * XSTR + n] = (_Float16)v;   // same rows, same wave: in-order DS
    }
  }

  // ---- Heads: h2(K=128) @ Whead^T (16 rows: 10 class + 2 action + 4 pad) -
  v8f acch;
#pragma unroll
  for (int i = 0; i < 8; ++i) acch[i] = 0.f;
  for (int k0 = 0; k0 < HID; k0 += 32) {
    v16h a = make_afrag(&xs[mrow * XSTR + k0 + aKoff]);
    acch = wmma_acc(a, &Wheadh[nloc * HID + k0 + bKoff], acch);
  }
  {
    int n = nloc;
    float bv = bhead[n];
#pragma unroll
    for (int r = 0; r < 8; ++r) {
      int mg = blockM + (wave << 4) + r + mAdd;
      float v = acch[r] + bv;
      if (n < 10) {
        out[(size_t)mg * 10 + n] = v;                       // class_pred
      } else if (n < 12) {
        out[(size_t)Btot * 10 + (size_t)mg * 2 + (n - 10)] = v;  // action_pred
      }
    }
  }
}

// ---------------------------------------------------------------------------
extern "C" void kernel_launch(void* const* d_in, const int* in_sizes, int n_in,
                              void* d_out, int out_size, void* d_ws,
                              size_t ws_size, hipStream_t stream) {
  const float* image = (const float*)d_in[0];
  const int* center  = (const int*)d_in[1];
  const float* h0    = (const float*)d_in[2];
  const float* Wih0  = (const float*)d_in[3];
  const float* bih0  = (const float*)d_in[4];
  const float* Whh0  = (const float*)d_in[5];
  const float* bhh0  = (const float*)d_in[6];
  const float* Wih1  = (const float*)d_in[7];
  const float* bih1  = (const float*)d_in[8];
  const float* Whh1  = (const float*)d_in[9];
  const float* bhh1  = (const float*)d_in[10];
  const float* Wcls  = (const float*)d_in[11];
  const float* bcls  = (const float*)d_in[12];
  const float* Wact  = (const float*)d_in[13];
  const float* bact  = (const float*)d_in[14];
  float* out = (float*)d_out;

  const int B = in_sizes[0] / 784;

  // Carve workspace (256B aligned slabs).
  size_t off = 0;
  auto carve = [&](size_t bytes) {
    void* p = (char*)d_ws + off;
    off = (off + bytes + 255) & ~(size_t)255;
    return p;
  };
  _Float16* Wih0h  = (_Float16*)carve((size_t)HID * KPAD * 2);
  _Float16* Whh0h  = (_Float16*)carve((size_t)HID * HID * 2);
  _Float16* Wih1h  = (_Float16*)carve((size_t)HID * HID * 2);
  _Float16* Whh1h  = (_Float16*)carve((size_t)HID * HID * 2);
  _Float16* Wheadh = (_Float16*)carve((size_t)16 * HID * 2);
  float* bias0 = (float*)carve(HID * 4);
  float* bias1 = (float*)carve(HID * 4);
  float* bhead = (float*)carve(16 * 4);
  _Float16* h0h = (_Float16*)carve((size_t)2 * B * HID * 2);

  prep_params<<<64, 256, 0, stream>>>(Wih0, Whh0, Wih1, Whh1, Wcls, Wact,
                                      bih0, bhh0, bih1, bhh1, bcls, bact,
                                      Wih0h, Whh0h, Wih1h, Whh1h, Wheadh,
                                      bias0, bias1, bhead);
  const int nH0 = 2 * B * HID;
  prep_h0<<<2048, 256, 0, stream>>>(h0, h0h, nH0);
  rnn_fused<<<B / BM, 128, 0, stream>>>(image, center, h0h, Wih0h, Whh0h,
                                        Wih1h, Whh1h, Wheadh, bias0, bias1,
                                        bhead, out, B);
}
